// Codebook_70128226009485
// MI455X (gfx1250) — matrix operand order repaired and checked by reading, not compiled, not run on hardware
//
#include <hip/hip_runtime.h>
#include <stdint.h>

#define B_   32
#define D_   384
#define T_   4096
#define K_   256
#define TT   128          // t-tile per block
#define DC   64           // d-chunk
#define NDC  (D_/DC)      // 6
#define XSP  72           // padded x row (u16): 144B row stride, 16B-aligned frags
#define CSP  72           // padded cb row (u16): 144B row stride, 16B-aligned frags
#define NT_  512          // threads per block (16 waves)
#define BETA 0.25f

typedef __attribute__((ext_vector_type(16))) __bf16        v16bf;
typedef __attribute__((ext_vector_type(8)))  float         v8f;
typedef __attribute__((ext_vector_type(4)))  float         f4;
typedef __attribute__((ext_vector_type(4)))  unsigned int  u4;

union Frag { unsigned int u[8]; u4 x[2]; v16bf v; };

__device__ __forceinline__ unsigned short f2bf(float f) {
    unsigned int u = __float_as_uint(f);
    u += 0x7FFFu + ((u >> 16) & 1u);   // round-to-nearest-even
    return (unsigned short)(u >> 16);
}

// ---------------- prep: codebook -> bf16, e2[k] = sum(e^2) ----------------
__global__ __launch_bounds__(256) void vq_prep(const float* __restrict__ cb,
                                               unsigned short* __restrict__ cbb,
                                               float* __restrict__ e2) {
    int k = threadIdx.x;                       // 256 threads == K_
    const float* row = cb + (size_t)k * D_;
    unsigned short* orow = cbb + (size_t)k * D_;
    float acc = 0.f;
    for (int d = 0; d < D_; ++d) {
        float v = row[d];
        orow[d] = f2bf(v);
        acc += v * v;
    }
    e2[k] = acc;
}

// ---------------- main: distances via WMMA, argmin, gather, loss partials ----------------
__global__ __launch_bounds__(NT_) void vq_main(const float* __restrict__ z,
                                               const float* __restrict__ cb,
                                               const unsigned short* __restrict__ cbb,
                                               const float* __restrict__ e2,
                                               float* __restrict__ partials,
                                               float* __restrict__ outq,
                                               float* __restrict__ outidx) {
    __shared__ __align__(16) unsigned short xs[TT][XSP];   // x tile, bf16 [t][d]
    __shared__ __align__(16) unsigned short cs[K_][CSP];   // codebook chunk, bf16 [k][d]
    __shared__ float e2s[K_];
    __shared__ float cand_v[TT][2];                        // per-(t, N-half) min value
    __shared__ int   cand_i[TT][2];                        // per-(t, N-half) min index
    __shared__ int   idx_sh[TT];
    __shared__ float wsum[16];

    const int tid   = threadIdx.x;
    const int lane  = tid & 31;
    const int wave  = tid >> 5;            // 0..15
    const int n     = lane & 15;           // column within 16-lane half
    const int half  = lane >> 4;
    const int bIdx  = blockIdx.x;
    const int b     = bIdx >> 5;           // 32 t-tiles per batch
    const int t0    = (bIdx & 31) * TT;
    const int mblk  = wave & 7;            // wave's M-block
    const int tbase = mblk * 16;
    const int kt0   = (wave >> 3) * 8;     // wave's N-half: k-tiles kt0..kt0+7

    const float* zb = z + (size_t)b * D_ * T_;

    if (tid < K_) e2s[tid] = e2[tid];

    v8f zro = {0.f,0.f,0.f,0.f,0.f,0.f,0.f,0.f};
    v8f c[8];
#pragma unroll
    for (int kt = 0; kt < 8; ++kt) c[kt] = zro;

    const int t4s = (tid & 31) * 4;        // this thread's fixed 4-t group
    const int dps = tid >> 5;              // 0..15

    for (int dc = 0; dc < NDC; ++dc) {
        const int d0 = dc * DC;
        __syncthreads();   // previous chunk's fragment reads done before overwrite

        // --- stage x chunk: 2 d-rows x 4 t per step, vector loads + packed stores ---
#pragma unroll
        for (int r = 0; r < 2; ++r) {
            int d = (dps + r * 16) * 2;                      // even, 0..62
            const f4 v0 = *(const f4*)&zb[(size_t)(d0 + d)     * T_ + t0 + t4s];
            const f4 v1 = *(const f4*)&zb[(size_t)(d0 + d + 1) * T_ + t0 + t4s];
            *(unsigned int*)&xs[t4s + 0][d] = (unsigned)f2bf(v0.x) | ((unsigned)f2bf(v1.x) << 16);
            *(unsigned int*)&xs[t4s + 1][d] = (unsigned)f2bf(v0.y) | ((unsigned)f2bf(v1.y) << 16);
            *(unsigned int*)&xs[t4s + 2][d] = (unsigned)f2bf(v0.z) | ((unsigned)f2bf(v1.z) << 16);
            *(unsigned int*)&xs[t4s + 3][d] = (unsigned)f2bf(v0.w) | ((unsigned)f2bf(v1.w) << 16);
        }
        // prefetch next d-chunk rows of z
        if (dc + 1 < NDC && tid < DC)
            __builtin_prefetch(zb + (size_t)(d0 + DC + tid) * T_ + t0, 0, 0);

        // --- stage codebook chunk via CDNA5 async global->LDS (b128 = 8 bf16) ---
        for (int i = tid; i < (K_ * DC / 8); i += NT_) {
            int k = i >> 3;               // DC/8 = 8 ops per row
            int j = (i & 7) * 8;          // element offset within row
            unsigned long long ga = (unsigned long long)(uintptr_t)(cbb + (size_t)k * D_ + d0 + j);
            unsigned int lo = (unsigned int)(uintptr_t)&cs[k][j];
            asm volatile("global_load_async_to_lds_b128 %0, %1, off"
                         :: "v"(lo), "v"(ga) : "memory");
        }
        asm volatile("s_wait_asynccnt 0" ::: "memory");
        __syncthreads();

        // --- GEMM: two 32-deep WMMA steps; all 8 B fragments live at once so the
        //     in-order DS returns allow partial s_wait_dscnt overlap with WMMA ---
#pragma unroll
        for (int s = 0; s < 2; ++s) {
            const int dk0 = s * 32;
            Frag a;                                    // 16-bit A 16x32 (ISA 7.12.2)
            {
                const unsigned short* pa = &xs[tbase + n][dk0 + 8 * half];
                a.x[0] = *(const u4*)pa;               // K pairs 0..7  (+8*half)
                a.x[1] = *(const u4*)(pa + 16);        // K pairs 16..23 (+8*half)
            }
            Frag bf[8];
#pragma unroll
            for (int kt = 0; kt < 8; ++kt) {
                const unsigned short* pb = &cs[(kt0 + kt) * 16 + n][dk0 + 8 * half];
                bf[kt].x[0] = *(const u4*)pb;
                bf[kt].x[1] = *(const u4*)(pb + 16);
            }
#pragma unroll
            for (int kt = 0; kt < 8; ++kt) {
                c[kt] = __builtin_amdgcn_wmma_f32_16x16x32_bf16(
                            false, a.v, false, bf[kt].v, (short)0, c[kt], false, false);
            }
        }
    }

    // --- argmin over k of (e2[k] - 2*S[t,k]); x^2 constant per t, irrelevant ---
    float best[8]; int bidx[8];
#pragma unroll
    for (int r = 0; r < 8; ++r) { best[r] = 3.4e38f; bidx[r] = 0; }
#pragma unroll
    for (int kt = 0; kt < 8; ++kt) {
        float ek = e2s[(kt0 + kt) * 16 + n];
#pragma unroll
        for (int r = 0; r < 8; ++r) {
            float v = ek - 2.0f * c[kt][r];    // lane-local k ascending -> first-min kept
            if (v < best[r]) { best[r] = v; bidx[r] = (kt0 + kt) * 16 + n; }
        }
    }
#pragma unroll
    for (int off = 8; off >= 1; off >>= 1) {   // reduce within each 16-lane half
#pragma unroll
        for (int r = 0; r < 8; ++r) {
            float ov = __shfl_xor(best[r], off, 32);
            int   oi = __shfl_xor(bidx[r], off, 32);
            if (ov < best[r] || (ov == best[r] && oi < bidx[r])) { best[r] = ov; bidx[r] = oi; }
        }
    }
    if (n == 0) {
        const int nh = wave >> 3;
#pragma unroll
        for (int r = 0; r < 8; ++r) {
            cand_v[tbase + 8 * half + r][nh] = best[r];
            cand_i[tbase + 8 * half + r][nh] = bidx[r];
        }
    }
    __syncthreads();

    // --- merge the two N-half candidates per t (ties -> lower index, matches argmin) ---
    if (tid < TT) {
        float v0 = cand_v[tid][0], v1 = cand_v[tid][1];
        int   i0 = cand_i[tid][0], i1 = cand_i[tid][1];
        int   ix = (v1 < v0 || (v1 == v0 && i1 < i0)) ? i1 : i0;
        idx_sh[tid] = ix;
        outidx[(size_t)b * T_ + t0 + tid] = (float)ix;
    }
    __syncthreads();

    // --- epilogue: straight-through output (== gathered code) + loss partials ---
    float lacc = 0.f;
    {
        const float* c0 = cb + (size_t)idx_sh[t4s + 0] * D_;
        const float* c1 = cb + (size_t)idx_sh[t4s + 1] * D_;
        const float* c2 = cb + (size_t)idx_sh[t4s + 2] * D_;
        const float* c3 = cb + (size_t)idx_sh[t4s + 3] * D_;
        for (int d = dps; d < D_; d += 16) {
            size_t off = (size_t)b * D_ * T_ + (size_t)d * T_ + t0 + t4s;
            f4 xv = __builtin_nontemporal_load((const f4*)&z[off]);   // last use of z tile
            f4 qv;
            qv.x = c0[d]; qv.y = c1[d]; qv.z = c2[d]; qv.w = c3[d];   // fp32 codebook, L2-hot
            __builtin_nontemporal_store(qv, (f4*)&outq[off]);         // x + sg(q-x) == q
            f4 df = qv - xv;
            lacc += df.x * df.x + df.y * df.y + df.z * df.z + df.w * df.w;
        }
    }
#pragma unroll
    for (int off = 16; off >= 1; off >>= 1) lacc += __shfl_xor(lacc, off, 32);
    if (lane == 0) wsum[wave] = lacc;
    __syncthreads();
    if (tid == 0) {
        float s = 0.f;
#pragma unroll
        for (int w = 0; w < 16; ++w) s += wsum[w];
        partials[bIdx] = s;          // deterministic per-block partial (no float atomics)
    }
}

// ---------------- deterministic loss reduction ----------------
__global__ __launch_bounds__(256) void vq_loss(const float* __restrict__ partials,
                                               float* __restrict__ loss) {
    __shared__ float s[256];
    float v = 0.f;
    for (int i = threadIdx.x; i < 1024; i += 256) v += partials[i];
    s[threadIdx.x] = v;
    __syncthreads();
    for (int w = 128; w > 0; w >>= 1) {
        if ((int)threadIdx.x < w) s[threadIdx.x] += s[threadIdx.x + w];
        __syncthreads();
    }
    if (threadIdx.x == 0)
        *loss = s[0] * (BETA / (float)((size_t)B_ * D_ * T_));
}

extern "C" void kernel_launch(void* const* d_in, const int* in_sizes, int n_in,
                              void* d_out, int out_size, void* d_ws, size_t ws_size,
                              hipStream_t stream) {
    const float* z  = (const float*)d_in[0];   // [B, D, T] fp32
    const float* cb = (const float*)d_in[1];   // [K, D]    fp32

    unsigned short* cbb = (unsigned short*)d_ws;                      // K*D bf16   (192 KB)
    float* e2       = (float*)((char*)d_ws + (size_t)K_ * D_ * 2);    // K floats
    float* partials = e2 + K_;                                        // 1024 floats

    float* outq   = (float*)d_out;                       // [B, D, T]
    float* outidx = outq + (size_t)B_ * D_ * T_;         // [B, T] (as float)
    float* loss   = outidx + (size_t)B_ * T_;            // scalar

    vq_prep<<<1, 256, 0, stream>>>(cb, cbb, e2);
    vq_main<<<B_ * (T_ / TT), NT_, 0, stream>>>(z, cb, cbb, e2, partials, outq, outidx);
    vq_loss<<<1, 256, 0, stream>>>(partials, loss);
}
